// BatchelorFwd_43301860278791
// MI455X (gfx1250) — matrix operand order repaired and checked by reading, not compiled, not run on hardware
//
#include <hip/hip_runtime.h>

typedef __attribute__((ext_vector_type(16))) _Float16 v16h;
typedef __attribute__((ext_vector_type(8)))  _Float16 v8h;
typedef __attribute__((ext_vector_type(8)))  float    v8f;

#define NN 256
#define NT 24
#define NC 16
#define LSTR 272          // LDS row stride in halves (256 + 16 pad, keeps 32B align)
#define PLANE 65536

#define WMMA(a, b, c) __builtin_amdgcn_wmma_f32_16x16x32_f16( \
    false, (a), false, (b), (short)0, (c), false, false)

// Load a 16x32 f16 A-fragment from row-major G storage.
// Lane layout (wave32): row m = m0 + lane%16, halves 0..15 = K x0+16h..+15
// (h = lane/16).
__device__ __forceinline__ v16h load_a_frag(const _Float16* g, int m, int x0, int h) {
  const _Float16* base = g + m * NN + x0 + 16 * h;
  v8h lo = *(const v8h*)(base);
  v8h hi = *(const v8h*)(base + 8);
  v16h r;
#pragma unroll
  for (int i = 0; i < 8; ++i) { r[i] = lo[i]; r[i + 8] = hi[i]; }
  return r;
}

// Build G[k,n] = exp(-2*pi*i*((k+128)(n+128) mod 256)/256) / 16 (f16 planes),
// plus a pre-negated imaginary plane so the GEMM needs no VALU negation.
__global__ void build_g_kernel(_Float16* g_r, _Float16* g_i, _Float16* g_in) {
  int p = blockIdx.x * blockDim.x + threadIdx.x;      // 0..65535
  int k = p >> 8, n = p & 255;
  int a = (k + 128) & 255, b = (n + 128) & 255;
  int pm = (a * b) & 255;
  float th = 0.024543692606f * (float)pm;             // 2*pi/256
  float ci = -__sinf(th) * 0.0625f;
  g_r[p]  = (_Float16)(__cosf(th) * 0.0625f);
  g_i[p]  = (_Float16)ci;
  g_in[p] = (_Float16)(-ci);
}

// Bilinear warp per frame; store f16 TRANSPOSED as [t][y][x] for B-fragment loads.
__global__ void warp_kernel(const float* __restrict__ imr, const float* __restrict__ imi,
                            const float* __restrict__ flow,
                            _Float16* __restrict__ owr, _Float16* __restrict__ owi) {
  int p = blockIdx.x * blockDim.x + threadIdx.x;      // 0..65535
  int t = blockIdx.y;
  int x = p >> 8, y = p & 255;
  int fb = (((x << 8) + y) * 2) * NT + t;             // flow[x,y,d,t]
  float X = fminf(fmaxf((float)x + flow[fb], 0.f), 255.f);
  float Y = fminf(fmaxf((float)y + flow[fb + NT], 0.f), 255.f);
  int x0 = (int)floorf(X); x0 = x0 < 0 ? 0 : (x0 > 254 ? 254 : x0);
  int y0 = (int)floorf(Y); y0 = y0 < 0 ? 0 : (y0 > 254 ? 254 : y0);
  float wx = X - (float)x0, wy = Y - (float)y0;
  int i00 = (x0 << 8) + y0;
  float w00 = (1.f - wx) * (1.f - wy), w01 = (1.f - wx) * wy;
  float w10 = wx * (1.f - wy),         w11 = wx * wy;
  float vr = w00 * imr[i00] + w01 * imr[i00 + 1] + w10 * imr[i00 + 256] + w11 * imr[i00 + 257];
  float vi = w00 * imi[i00] + w01 * imi[i00 + 1] + w10 * imi[i00 + 256] + w11 * imi[i00 + 257];
  int o = t * PLANE + (y << 8) + x;                   // [t][y][x]
  owr[o] = (_Float16)vr;
  owi[o] = (_Float16)vi;
}

// f16 coil maps, transposed to [c][y][x]
__global__ void stage_smaps_kernel(const float* __restrict__ smr, const float* __restrict__ smi,
                                   _Float16* __restrict__ osr, _Float16* __restrict__ osi) {
  int p = blockIdx.x * blockDim.x + threadIdx.x;
  int c = blockIdx.y;
  int x = p >> 8, y = p & 255;
  int src = (((x << 8) + y) * NC) + c;                // smaps[x,y,c]
  int o = c * PLANE + (y << 8) + x;
  osr[o] = (_Float16)smr[src];
  osi[o] = (_Float16)smi[src];
}

// One workgroup = one (t,c): full 2-D FFT via two complex GEMMs with G,
// P1 held in LDS (f16, 272-half stride, 278528 B), masked atomic accumulate.
// Each wave owns a fixed n-block, so its B fragments are loop-invariant and
// get hoisted into registers; G A-fragments (3 planes incl. -Gi) stream from L2.
__global__ __launch_bounds__(512) void fft2_wmma_accum_kernel(
    const _Float16* __restrict__ g_r, const _Float16* __restrict__ g_i,
    const _Float16* __restrict__ g_in,
    const _Float16* __restrict__ wrp_r, const _Float16* __restrict__ wrp_i,
    const _Float16* __restrict__ smp_r, const _Float16* __restrict__ smp_i,
    const float* __restrict__ mask, float* __restrict__ out) {
  extern __shared__ _Float16 lds[];                   // [2][256][LSTR]
  _Float16* p1r = lds;
  _Float16* p1i = lds + NN * LSTR;

  int t = blockIdx.x % NT;
  int c = blockIdx.x / NT;
  int lane = threadIdx.x & 31;
  int wave = threadIdx.x >> 5;                        // 16 waves
  int h = lane >> 4, l16 = lane & 15;

  const _Float16* wr = wrp_r + t * PLANE;
  const _Float16* wi = wrp_i + t * PLANE;
  const _Float16* sr = smp_r + c * PLANE;
  const _Float16* si = smp_i + c * PLANE;

  // ---------------- Phase 1: P1[kx, y] = sum_x G[kx,x] * coil[x,y] ----------------
  for (int tile = wave; tile < 256; tile += 16) {
    int m0 = (tile >> 4) << 4;                        // kx block (varies per wave)
    int n0 = (tile & 15) << 4;                        // y block (fixed per wave)
    v8f accr = {}; v8f acci = {};
    int m = m0 + l16;
    int y = n0 + l16;
    for (int x0 = 0; x0 < NN; x0 += 32) {
      v16h agr  = load_a_frag(g_r,  m, x0, h);
      v16h agi  = load_a_frag(g_i,  m, x0, h);
      v16h agin = load_a_frag(g_in, m, x0, h);
      // B: on-the-fly coil, 16 contiguous x at row y (32B aligned); invariant
      // over the tile loop -> hoisted by the compiler.
      int boff = y * NN + x0 + 16 * h;
      v16h bwr = *(const v16h*)(wr + boff);
      v16h bwi = *(const v16h*)(wi + boff);
      v16h bsr = *(const v16h*)(sr + boff);
      v16h bsi = *(const v16h*)(si + boff);
      v16h bcr = bwr * bsr - bwi * bsi;               // coil real (packed f16)
      v16h bci = bwr * bsi + bwi * bsr;               // coil imag
      accr = WMMA(agr,  bcr, accr);
      accr = WMMA(agin, bci, accr);
      acci = WMMA(agr,  bci, acci);
      acci = WMMA(agi,  bcr, acci);
    }
    int mb = m0 + 8 * h;                              // C/D layout: M = v + 8h, N = l16
#pragma unroll
    for (int v = 0; v < 8; ++v) {
      int addr = (mb + v) * LSTR + n0 + l16;          // P1 row-major [kx][y]
      p1r[addr] = (_Float16)accr[v];
      p1i[addr] = (_Float16)acci[v];
    }
  }
  __syncthreads();

  // ------------- Phase 2: P2[ky, kx] = sum_y G[ky,y] * P1[kx, y] = Y[kx,ky] -------------
  for (int tile = wave; tile < 256; tile += 16) {
    int m0 = (tile >> 4) << 4;                        // ky block
    int n0 = (tile & 15) << 4;                        // kx block (fixed per wave)
    v8f accr = {}; v8f acci = {};
    int m = m0 + l16;
    int kx = n0 + l16;
    for (int y0 = 0; y0 < NN; y0 += 32) {
      v16h agr  = load_a_frag(g_r,  m, y0, h);
      v16h agi  = load_a_frag(g_i,  m, y0, h);
      v16h agin = load_a_frag(g_in, m, y0, h);
      int boff = kx * LSTR + y0 + 16 * h;             // contiguous y, 32B aligned
      v16h bpr = *(const v16h*)(p1r + boff);
      v16h bpi = *(const v16h*)(p1i + boff);
      accr = WMMA(agr,  bpr, accr);
      accr = WMMA(agin, bpi, accr);
      acci = WMMA(agr,  bpi, acci);
      acci = WMMA(agi,  bpr, acci);
    }
    int kyb = m0 + 8 * h;
#pragma unroll
    for (int v = 0; v < 8; ++v) {
      int ky = kyb + v;
      int pix = (kx << 8) + ky;                       // Y[kx,ky] = P2[ky,kx]
      float mv = mask[(pix * NC + c) * NT + t];       // mask[kx,ky,c,t]
      atomicAdd(out + pix * NC + c,           mv * accr[v]);   // real plane
      atomicAdd(out + 1048576 + pix * NC + c, mv * acci[v]);   // imag plane
    }
  }
}

extern "C" void kernel_launch(void* const* d_in, const int* in_sizes, int n_in,
                              void* d_out, int out_size, void* d_ws, size_t ws_size,
                              hipStream_t stream) {
  const float* image_real = (const float*)d_in[0];
  const float* image_imag = (const float*)d_in[1];
  const float* mask       = (const float*)d_in[2];
  const float* smaps_real = (const float*)d_in[3];
  const float* smaps_imag = (const float*)d_in[4];
  const float* flow       = (const float*)d_in[5];
  float* out = (float*)d_out;

  // Workspace (f16 halves): G(3 planes) | warped(24x2) | smaps(16x2) ~= 10.9 MB
  _Float16* ws   = (_Float16*)d_ws;
  _Float16* g_r  = ws;
  _Float16* g_i  = g_r + PLANE;
  _Float16* g_in = g_i + PLANE;
  _Float16* w_r  = g_in + PLANE;
  _Float16* w_i  = w_r + NT * PLANE;
  _Float16* s_r  = w_i + NT * PLANE;
  _Float16* s_i  = s_r + NC * PLANE;

  // Output accumulated with atomics across t -> must re-zero every call.
  hipMemsetAsync(d_out, 0, (size_t)out_size * sizeof(float), stream);

  build_g_kernel<<<256, 256, 0, stream>>>(g_r, g_i, g_in);
  warp_kernel<<<dim3(256, NT), 256, 0, stream>>>(image_real, image_imag, flow, w_r, w_i);
  stage_smaps_kernel<<<dim3(256, NC), 256, 0, stream>>>(smaps_real, smaps_imag, s_r, s_i);

  size_t lds_bytes = (size_t)2 * NN * LSTR * sizeof(_Float16);   // 278528 B < 320 KB WGP LDS
  fft2_wmma_accum_kernel<<<NT * NC, 512, lds_bytes, stream>>>(
      g_r, g_i, g_in, w_r, w_i, s_r, s_i, mask, out);
}
